// AttentionHead_26362509263329
// MI455X (gfx1250) — matrix operand order repaired and checked by reading, not compiled, not run on hardware
//
#include <hip/hip_runtime.h>
#include <hip/hip_bf16.h>

typedef __attribute__((ext_vector_type(16))) __bf16 v16bf;
typedef __attribute__((ext_vector_type(8)))  __bf16 v8bf;
typedef __attribute__((ext_vector_type(8)))  float  v8f;

#define D_MODEL 1024
#define N_SEQ   4096

// ---------- helpers ----------

__device__ __forceinline__ unsigned short f2bf_bits(float f) {
    unsigned u = __builtin_bit_cast(unsigned, f);
    u += 0x7FFFu + ((u >> 16) & 1u);   // round-to-nearest-even
    return (unsigned short)(u >> 16);
}
__device__ __forceinline__ __bf16 f2bf16(float f) {
    unsigned short s = f2bf_bits(f);
    return __builtin_bit_cast(__bf16, s);
}

__device__ __forceinline__ v8f wmma_bf16(v16bf a, v16bf b, v8f c) {
    return __builtin_amdgcn_wmma_f32_16x16x32_bf16(
        /*neg_a=*/false, a, /*neg_b=*/false, b,
        /*c_mod=*/(short)0, c, /*reuse_a=*/false, /*reuse_b=*/false);
}

// A fragment: 16x32 bf16 tile of row-major A (lda elements per row).
// ISA layout: lane holds row (lane&15); kb=(lane>>4)*8; elems 0..7 -> K=k0+kb..+7,
// elems 8..15 -> K=k0+16+kb..+7.
__device__ __forceinline__ v16bf load_a_frag(const __bf16* A, int row0, int lda,
                                             int k0, int lane) {
    int r  = row0 + (lane & 15);
    int kb = (lane >> 4) * 8;
    const __bf16* p = A + (size_t)r * lda + k0 + kb;
    v8bf lo = *(const v8bf*)p;
    v8bf hi = *(const v8bf*)(p + 16);
    return __builtin_shufflevector(lo, hi, 0,1,2,3,4,5,6,7,8,9,10,11,12,13,14,15);
}

// B fragment: 32x16 tile; B stored transposed row-major: Bt[n][k], ldb elements/row.
// ISA layout: lane holds col (lane&15); kb=(lane>>4)*8; elems 0..7 -> K=k0+kb..+7,
// elems 8..15 -> K=k0+16+kb..+7.
__device__ __forceinline__ v16bf load_b_frag(const __bf16* Bt, int col0, int ldb,
                                             int k0, int lane) {
    int n  = col0 + (lane & 15);
    int kk = k0 + (lane >> 4) * 8;
    const __bf16* p = Bt + (size_t)n * ldb + kk;
    v8bf lo = *(const v8bf*)p;
    v8bf hi = *(const v8bf*)(p + 16);
    return __builtin_shufflevector(lo, hi, 0,1,2,3,4,5,6,7,8,9,10,11,12,13,14,15);
}

// ---------- prep: fp32 -> bf16 copies / transposes ----------

__global__ void prep_x_kernel(const float* __restrict__ x,
                              __bf16* __restrict__ Xb,
                              __bf16* __restrict__ XbT) {
    int idx = blockIdx.x * 256 + threadIdx.x;       // over N_SEQ*D_MODEL
    int i = idx >> 10, d = idx & 1023;
    __bf16 v = f2bf16(x[idx]);
    Xb[idx] = v;
    XbT[(size_t)d * N_SEQ + i] = v;
}

__global__ void prep_w_kernel(const float* __restrict__ wqk,
                              const float* __restrict__ wov,
                              __bf16* __restrict__ WqkT,
                              __bf16* __restrict__ WovT) {
    int idx = blockIdx.x * 256 + threadIdx.x;       // over D*D
    int r = idx >> 10, c = idx & 1023;
    WqkT[(size_t)c * D_MODEL + r] = f2bf16(wqk[idx]);
    WovT[(size_t)c * D_MODEL + r] = f2bf16(wov[idx]);
}

// ---------- generic bf16 GEMM: C = A(row-major MxK) * Bt^T (Bt is NxK) ----------
// Block: 256 thr = 8 waves; block tile 64x256; wave tile 32x64 (2x4 WMMA tiles).
// 8 WMMAs per k-chunk against 6 fragment loads -> ~1.33x the FLOPs/byte of a 2x2 tile.

template<bool OUT_BF16>
__global__ __launch_bounds__(256)
void gemm_kernel(const __bf16* __restrict__ A, const __bf16* __restrict__ Bt,
                 void* __restrict__ Cout, int K, int ldc) {
    const int wave = threadIdx.x >> 5;
    const int lane = threadIdx.x & 31;
    const int bm = blockIdx.x * 64  + (wave >> 2) * 32;   // 2 wave rows
    const int bn = blockIdx.y * 256 + (wave & 3) * 64;    // 4 wave cols

    v8f acc[2][4];
    for (int i = 0; i < 2; ++i)
        for (int j = 0; j < 4; ++j) acc[i][j] = (v8f){};

    for (int k0 = 0; k0 < K; k0 += 32) {
        if (k0 + 128 < K) {   // prefetch the k+128 slab (global_prefetch_b8)
            __builtin_prefetch(A  + (size_t)(bm + (lane & 15)) * K + k0 + 128, 0, 1);
            __builtin_prefetch(Bt + (size_t)(bn + (lane & 15)) * K + k0 + 128, 0, 1);
        }
        v16bf a0 = load_a_frag(A, bm,      K, k0, lane);
        v16bf a1 = load_a_frag(A, bm + 16, K, k0, lane);
        v16bf b0 = load_b_frag(Bt, bn,      K, k0, lane);
        v16bf b1 = load_b_frag(Bt, bn + 16, K, k0, lane);
        v16bf b2 = load_b_frag(Bt, bn + 32, K, k0, lane);
        v16bf b3 = load_b_frag(Bt, bn + 48, K, k0, lane);
        acc[0][0] = wmma_bf16(a0, b0, acc[0][0]);
        acc[1][0] = wmma_bf16(a1, b0, acc[1][0]);
        acc[0][1] = wmma_bf16(a0, b1, acc[0][1]);
        acc[1][1] = wmma_bf16(a1, b1, acc[1][1]);
        acc[0][2] = wmma_bf16(a0, b2, acc[0][2]);
        acc[1][2] = wmma_bf16(a1, b2, acc[1][2]);
        acc[0][3] = wmma_bf16(a0, b3, acc[0][3]);
        acc[1][3] = wmma_bf16(a1, b3, acc[1][3]);
    }

    const int half = lane >> 4, n = lane & 15;
    for (int i = 0; i < 2; ++i)
        for (int j = 0; j < 4; ++j) {
            int col = bn + j * 16 + n;
            for (int v = 0; v < 8; ++v) {
                int row = bm + i * 16 + v + half * 8;
                float val = acc[i][j][v];
                if (OUT_BF16)
                    ((__bf16*)Cout)[(size_t)row * ldc + col] = f2bf16(val);
                else
                    ((float*)Cout)[(size_t)row * ldc + col] = val;
            }
        }
}

// ---------- attention band kernel ----------
// One block = 16 query rows. S band (16 x 4096 f32) lives in LDS (256 KB).
// Phase 1: S = Q Xb^T (causal) via WMMA.  Phase 2: row softmax, repack P as
// bf16 in place.  Phase 3: O = P X via WMMA (A from LDS, B from XbT).

__global__ __launch_bounds__(256)
void attn_kernel(const __bf16* __restrict__ Qb, const __bf16* __restrict__ Xb,
                 const __bf16* __restrict__ XbT, __bf16* __restrict__ Ob) {
    extern __shared__ float sS[];                  // [16][N_SEQ]
    const int wave = threadIdx.x >> 5;
    const int lane = threadIdx.x & 31;
    const int half = lane >> 4;
    const int i0   = blockIdx.x * 16;
    const int ncols = i0 + 16;                     // causal columns
    const int ntiles = ncols >> 4;

    // ---- phase 1: score tiles (pairs of tiles share one Q fragment) ----
    int t = wave;
    for (; t + 8 < ntiles; t += 16) {
        int j0 = t * 16, j1 = (t + 8) * 16;
        v8f acc0 = {}, acc1 = {};
        for (int k0 = 0; k0 < D_MODEL; k0 += 32) {
            v16bf a  = load_a_frag(Qb, i0, D_MODEL, k0, lane);
            v16bf b0 = load_b_frag(Xb, j0, D_MODEL, k0, lane);
            v16bf b1 = load_b_frag(Xb, j1, D_MODEL, k0, lane);
            acc0 = wmma_bf16(a, b0, acc0);
            acc1 = wmma_bf16(a, b1, acc1);
        }
        int c0 = j0 + (lane & 15), c1 = j1 + (lane & 15);
        for (int v = 0; v < 8; ++v) {
            int m = v + half * 8;
            sS[m * N_SEQ + c0] = ((i0 + m) >= c0) ? acc0[v] : -__builtin_inff();
            sS[m * N_SEQ + c1] = ((i0 + m) >= c1) ? acc1[v] : -__builtin_inff();
        }
    }
    if (t < ntiles) {                              // leftover single tile
        int j0 = t * 16;
        v8f acc = {};
        for (int k0 = 0; k0 < D_MODEL; k0 += 32) {
            v16bf a = load_a_frag(Qb, i0, D_MODEL, k0, lane);
            v16bf b = load_b_frag(Xb, j0, D_MODEL, k0, lane);
            acc = wmma_bf16(a, b, acc);
        }
        int col = j0 + (lane & 15);
        for (int v = 0; v < 8; ++v) {
            int m = v + half * 8;
            sS[m * N_SEQ + col] = ((i0 + m) >= col) ? acc[v] : -__builtin_inff();
        }
    }
    __syncthreads();

    // ---- phase 2: softmax per row; pack P to bf16 in place ----
    const int padded = (ncols + 31) & ~31;
    for (int r = wave * 2; r < wave * 2 + 2; ++r) {
        float* row = sS + r * N_SEQ;
        float mx = -__builtin_inff();
        for (int c = lane; c < ncols; c += 32) mx = fmaxf(mx, row[c]);
        for (int off = 16; off; off >>= 1) mx = fmaxf(mx, __shfl_xor(mx, off, 32));
        float l = 0.f;
        for (int c = lane; c < ncols; c += 32) l += __expf(row[c] - mx);
        for (int off = 16; off; off >>= 1) l += __shfl_xor(l, off, 32);
        float inv = 1.0f / l;
        int npairs = padded >> 1;
        for (int p = lane; p < npairs; p += 32) {
            int c0 = 2 * p, c1 = 2 * p + 1;
            float e0 = (c0 < ncols) ? __expf(row[c0] - mx) * inv : 0.f;
            float e1 = (c1 < ncols) ? __expf(row[c1] - mx) * inv : 0.f;
            unsigned packed = (unsigned)f2bf_bits(e0) |
                              ((unsigned)f2bf_bits(e1) << 16);
            row[p] = __builtin_bit_cast(float, packed);   // same type, no TBAA issue
        }
    }
    __syncthreads();

    // ---- phase 3: O = P @ X ; wave handles 128 feature cols (8 WMMA tiles) ----
    const char* P = (const char*)sS;               // bf16 rows, 16 KB apart
    const int f_base = wave * 128;
    v8f o[8];
    for (int tt = 0; tt < 8; ++tt) o[tt] = (v8f){};
    const int kchunks = padded >> 5;
    const int r = lane & 15, kb = half * 8;
    for (int kc = 0; kc < kchunks; ++kc) {
        int K0 = kc * 32;
        const char* pr = P + (size_t)r * (N_SEQ * 4);
        v8bf lo = *(const v8bf*)(pr + 2 * (K0 + kb));
        v8bf hi = *(const v8bf*)(pr + 2 * (K0 + 16 + kb));
        v16bf a = __builtin_shufflevector(lo, hi,
                    0,1,2,3,4,5,6,7,8,9,10,11,12,13,14,15);
        for (int tt = 0; tt < 8; ++tt) {
            v16bf b = load_b_frag(XbT, f_base + tt * 16, N_SEQ, K0, lane);
            o[tt] = wmma_bf16(a, b, o[tt]);
        }
    }
    const int n = lane & 15;
    for (int tt = 0; tt < 8; ++tt) {
        int col = f_base + tt * 16 + n;
        for (int v = 0; v < 8; ++v) {
            int rowg = i0 + v + half * 8;
            Ob[(size_t)rowg * D_MODEL + col] = f2bf16(o[tt][v]);
        }
    }
}

// ---------- launcher ----------

extern "C" void kernel_launch(void* const* d_in, const int* in_sizes, int n_in,
                              void* d_out, int out_size, void* d_ws, size_t ws_size,
                              hipStream_t stream) {
    const float* x   = (const float*)d_in[0];
    const float* wqk = (const float*)d_in[1];
    const float* wov = (const float*)d_in[2];
    float* out = (float*)d_out;

    char* ws = (char*)d_ws;
    const size_t MB = 1024 * 1024;
    __bf16* Xb   = (__bf16*)(ws);                  //  8 MB  x bf16 row-major
    __bf16* XbT  = (__bf16*)(ws + 8  * MB);        //  8 MB  x^T bf16
    __bf16* Qb   = (__bf16*)(ws + 16 * MB);        //  8 MB  q bf16
    __bf16* Ob   = (__bf16*)(ws + 24 * MB);        //  8 MB  attn@x bf16
    __bf16* WqkT = (__bf16*)(ws + 32 * MB);        //  2 MB  wqk^T bf16
    __bf16* WovT = (__bf16*)(ws + 34 * MB);        //  2 MB  wov^T bf16

    prep_x_kernel<<<(N_SEQ * D_MODEL) / 256, 256, 0, stream>>>(x, Xb, XbT);
    prep_w_kernel<<<(D_MODEL * D_MODEL) / 256, 256, 0, stream>>>(wqk, wov, WqkT, WovT);

    // Q = x @ wqk   (4096x1024x1024), bf16 output
    gemm_kernel<true><<<dim3(N_SEQ / 64, D_MODEL / 256), 256, 0, stream>>>(
        Xb, WqkT, (void*)Qb, D_MODEL, D_MODEL);

    // causal attention bands: 16 query rows / block, 256 KB dynamic LDS
    attn_kernel<<<N_SEQ / 16, 256, 16 * N_SEQ * sizeof(float), stream>>>(
        Qb, Xb, XbT, Ob);

    // out = O @ wov (4096x1024x1024), f32 output
    gemm_kernel<false><<<dim3(N_SEQ / 64, D_MODEL / 256), 256, 0, stream>>>(
        Ob, WovT, (void*)out, D_MODEL, D_MODEL);
}
